// ViT_PMT_CLU_86775519248580
// MI455X (gfx1250) — matrix operand order repaired and checked by reading, not compiled, not run on hardware
//
#include <hip/hip_runtime.h>
#include <math.h>

typedef __bf16 bf16_t;
typedef float  v8f  __attribute__((ext_vector_type(8)));
typedef float  v2f  __attribute__((ext_vector_type(2)));
typedef bf16_t v16bf __attribute__((ext_vector_type(16)));

namespace {
constexpr int NTOK = 4096;   // tokens
constexpr int IND  = 1024;   // input dim
constexpr int E    = 512;    // embed dim
constexpr int H    = 8;      // heads
constexpr int DH   = 64;     // head dim
constexpr int MM   = 256;    // landmarks (M)
constexpr int NP   = 4352;   // padded seq len (both blocks)
constexpr int PAD1 = 247;    // block-1 left pad (4105 -> 4352)
constexpr int PAD0 = 255;    // block-0 left pad (4097 -> 4352)
constexpr int LL   = 17;     // n // M
constexpr int CC   = 8;      // clusters
constexpr int KWC  = 33;     // conv kernel width
constexpr float NEGV = -1.7014118e38f;  // -FLT_MAX/2
}

// ---------------------------------------------------------------------------
// block reduction helpers (blockDim.x == 256)
// ---------------------------------------------------------------------------
__device__ inline float block_reduce_sum(float v, float* red) {
    int tid = threadIdx.x;
    red[tid] = v; __syncthreads();
    for (int s = 128; s > 0; s >>= 1) { if (tid < s) red[tid] += red[tid + s]; __syncthreads(); }
    float r = red[0]; __syncthreads();
    return r;
}
__device__ inline float block_reduce_max(float v, float* red) {
    int tid = threadIdx.x;
    red[tid] = v; __syncthreads();
    for (int s = 128; s > 0; s >>= 1) { if (tid < s) red[tid] = fmaxf(red[tid], red[tid + s]); __syncthreads(); }
    float r = red[0]; __syncthreads();
    return r;
}

// ---------------------------------------------------------------------------
// bf16 WMMA GEMM: C = A(MxK) @ B(KxN) [+bias][+residual][relu]
// A row-major bf16.  B is ALWAYS passed transposed: Bt is (N x K) row-major,
// so every lane reads 16 contiguous bf16 (two global_load_b128).
// 4 waves / block; each wave computes a 16x32 tile (one A frag, two WMMAs).
// Optional f32 and bf16 outputs. Batched via blockIdx.z with element strides.
// ---------------------------------------------------------------------------
__global__ __launch_bounds__(128) void gemm_bf16_kernel(
    const bf16_t* __restrict__ A, const bf16_t* __restrict__ Bt,
    int M, int N, int K, int lda, int ldb,
    float* Cf, bf16_t* Cb, int ldc,
    const float* bias, const float* resid, int ldres, int relu,
    long long sA, long long sB, long long sC, long long sRes)
{
    int batch = blockIdx.z;
    const bf16_t* Ab = A  + (long long)batch * sA;
    const bf16_t* Bb = Bt + (long long)batch * sB;
    int wave = threadIdx.x >> 5;
    int lane = threadIdx.x & 31;
    int n0 = (blockIdx.x * 4 + wave) * 32;     // 32 output columns per wave
    int m0 = blockIdx.y * 16;
    if (n0 >= N || m0 >= M) return;
    int row = lane & 15;
    int hi  = lane >> 4;
    v8f acc0 = {}, acc1 = {};
    const bf16_t* Arow = Ab + (long long)(m0 + row) * lda;
    const bf16_t* B0   = Bb + (long long)(n0 + row) * ldb + hi * 16;
    const bf16_t* B1   = Bb + (long long)(n0 + 16 + row) * ldb + hi * 16;
    for (int k0 = 0; k0 < K; k0 += 32) {
        v16bf a, b0, b1;
#pragma unroll
        for (int j = 0; j < 16; ++j) {
            int kk = j + hi * 16;   // contiguous k = 16*hi + j (matches B)
            a[j] = Arow[k0 + kk];
        }
#pragma unroll
        for (int j = 0; j < 16; ++j) b0[j] = B0[k0 + j];
#pragma unroll
        for (int j = 0; j < 16; ++j) b1[j] = B1[k0 + j];
        __builtin_prefetch(Arow + k0 + 64, 0, 1);       // global_prefetch_b8
        __builtin_prefetch(B0 + k0 + 64, 0, 1);
        acc0 = __builtin_amdgcn_wmma_f32_16x16x32_bf16(false, a, false, b0, (short)0, acc0,
                                                       false, false);
        acc1 = __builtin_amdgcn_wmma_f32_16x16x32_bf16(false, a, false, b1, (short)0, acc1,
                                                       false, false);
    }
#pragma unroll
    for (int r = 0; r < 8; ++r) {
        int mm = m0 + r + hi * 8;
#pragma unroll
        for (int t = 0; t < 2; ++t) {
            int nn = n0 + t * 16 + row;
            float v = (t == 0) ? acc0[r] : acc1[r];
            if (bias)  v += bias[nn];
            if (resid) v += resid[(long long)batch * sRes + (long long)mm * ldres + nn];
            if (relu)  v = v > 0.f ? v : 0.f;
            long long o = (long long)batch * sC + (long long)mm * ldc + nn;
            if (Cf) Cf[o] = v;
            if (Cb) Cb[o] = (bf16_t)v;
        }
    }
}

// ---------------------------------------------------------------------------
// f32 WMMA GEMM (16x16x4):  out = dA*I + s*(A @ B)   (Newton-Schulz pinv)
// dual 16x16 tiles per wave as well
// ---------------------------------------------------------------------------
__global__ __launch_bounds__(128) void gemm_f32_kernel(
    const float* __restrict__ A, const float* __restrict__ B,
    int M, int N, int K, int lda, int ldb,
    float* Cf, int ldc, float dA, float s,
    long long sA, long long sB, long long sC)
{
    int batch = blockIdx.z;
    const float* Ab = A + (long long)batch * sA;
    const float* Bb = B + (long long)batch * sB;
    int wave = threadIdx.x >> 5;
    int lane = threadIdx.x & 31;
    int n0 = (blockIdx.x * 4 + wave) * 32;
    int m0 = blockIdx.y * 16;
    if (n0 >= N || m0 >= M) return;
    int row = lane & 15;
    int hi  = lane >> 4;
    v8f acc0 = {}, acc1 = {};
    const float* Arow = Ab + (long long)(m0 + row) * lda;
    for (int k0 = 0; k0 < K; k0 += 4) {
        v2f a, b0, b1;
        a.x = Arow[k0 + hi * 2];
        a.y = Arow[k0 + hi * 2 + 1];
        const float* Bp = Bb + (long long)(k0 + hi * 2) * ldb + row;
        b0.x = Bp[n0];
        b0.y = Bp[n0 + ldb];
        b1.x = Bp[n0 + 16];
        b1.y = Bp[n0 + 16 + ldb];
        acc0 = __builtin_amdgcn_wmma_f32_16x16x4_f32(false, a, false, b0, (short)0, acc0,
                                                     false, false);
        acc1 = __builtin_amdgcn_wmma_f32_16x16x4_f32(false, a, false, b1, (short)0, acc1,
                                                     false, false);
    }
#pragma unroll
    for (int r = 0; r < 8; ++r) {
        int mm = m0 + r + hi * 8;
#pragma unroll
        for (int t = 0; t < 2; ++t) {
            int nn = n0 + t * 16 + row;
            float v = dA * (mm == nn ? 1.f : 0.f) + s * ((t == 0) ? acc0[r] : acc1[r]);
            Cf[(long long)batch * sC + (long long)mm * ldc + nn] = v;
        }
    }
}

// ---------------------------------------------------------------------------
// small utility kernels
// ---------------------------------------------------------------------------
__global__ void cvt_kernel(const float* in, bf16_t* out, long long n) {
    for (long long t = blockIdx.x * 256LL + threadIdx.x; t < n;
         t += (long long)gridDim.x * 256) out[t] = (bf16_t)in[t];
}

// out[c][r] = (bf16) in[r][c]   (weights -> B^T layout, one-time)
__global__ void cvt_trans_kernel(const float* in, bf16_t* out, int R, int C) {
    long long n = (long long)R * C;
    for (long long t = blockIdx.x * 256LL + threadIdx.x; t < n;
         t += (long long)gridDim.x * 256) {
        int r = (int)(t / C), c = (int)(t % C);
        out[(long long)c * R + r] = (bf16_t)in[t];
    }
}

__global__ void build_seq1_kernel(const float* cls, const float* pmt, float* seq) {
    int r = blockIdx.x;                     // 0..8
    const float* src = (r == 0) ? cls : pmt + (long long)(r - 1) * E;
    float* dst = seq + (long long)(PAD1 + r) * E;
    for (int e = threadIdx.x; e < E; e += 256) dst[e] = src[e];
}

__global__ void set_mask1_kernel(float* mask) {
    for (int i = blockIdx.x * 256 + threadIdx.x; i < NP; i += gridDim.x * 256)
        mask[i] = (i >= PAD1) ? 1.f : 0.f;
}

__global__ void layernorm_kernel(const float* X, bf16_t* Y, const float* g,
                                 const float* b, int row0) {
    __shared__ float red[256];
    int r = blockIdx.x, tid = threadIdx.x;
    bf16_t* y = Y + (long long)r * E;
    if (r < row0) { for (int e = tid; e < E; e += 256) y[e] = (bf16_t)0.0f; return; }
    const float* x = X + (long long)r * E;
    float s = 0;
    for (int e = tid; e < E; e += 256) s += x[e];
    float mu = block_reduce_sum(s, red) * (1.f / E);
    float v = 0;
    for (int e = tid; e < E; e += 256) { float d = x[e] - mu; v += d * d; }
    float var = block_reduce_sum(v, red) * (1.f / E);
    float rstd = rsqrtf(var + 1e-5f);
    for (int e = tid; e < E; e += 256)
        y[e] = (bf16_t)((x[e] - mu) * rstd * g[e] + b[e]);
}

// split qkv; q scaled by DH^-0.5; v additionally written transposed (DH x NP)
__global__ void qkv_split_kernel(const float* qkv, const float* mask,
                                 float* qF, float* kF, float* vF,
                                 bf16_t* qB, bf16_t* kB, bf16_t* vTB) {
    long long n = (long long)H * NP * DH;
    for (long long t = blockIdx.x * 256LL + threadIdx.x; t < n;
         t += (long long)gridDim.x * 256) {
        int d = (int)(t % DH);
        long long r = t / DH;
        int i = (int)(r % NP);
        int h = (int)(r / NP);
        float m = mask[i];
        const float* src = qkv + (long long)i * (3 * E);
        float q = src[h * DH + d] * m * 0.125f;           // DH^-0.5
        float k = src[E + h * DH + d] * m;
        float v = src[2 * E + h * DH + d] * m;
        long long o = ((long long)h * NP + i) * DH + d;
        qF[o] = q; kF[o] = k; vF[o] = v;
        qB[o] = (bf16_t)q; kB[o] = (bf16_t)k;
        vTB[((long long)h * DH + d) * NP + i] = (bf16_t)v;
    }
}

__global__ void mls_kernel(const float* mask, float* mls) {
    int m = blockIdx.x * 256 + threadIdx.x;
    if (m >= MM) return;
    float s = 0;
    for (int j = 0; j < LL; ++j) s += mask[m * LL + j];
    mls[m] = s;
}

__global__ void landmark_kernel(const float* qF, const float* kF, const float* mls,
                                bf16_t* qlB, bf16_t* klB) {
    int m = blockIdx.x, h = blockIdx.y, d = threadIdx.x;
    float sq = 0, sk = 0;
    for (int j = 0; j < LL; ++j) {
        int n = m * LL + j;
        sq += qF[((long long)h * NP + n) * DH + d];
        sk += kF[((long long)h * NP + n) * DH + d];
    }
    float dv = mls[m] + 1e-8f;
    qlB[((long long)h * MM + m) * DH + d] = (bf16_t)(sq / dv);
    klB[((long long)h * MM + m) * DH + d] = (bf16_t)(sk / dv);
}

// masked row softmax; valid = rowmask[row]>0 && colmask[col]>0, else NEGV
__global__ void softmax_mask_kernel(const float* S, bf16_t* Pb, float* Pf,
                                    const float* rowmask, const float* colmask,
                                    int cols, int ldS, long long sS,
                                    long long sPb, long long sPf) {
    __shared__ float red[256];
    int h = blockIdx.y, r = blockIdx.x, tid = threadIdx.x;
    const float* Sr = S + (long long)h * sS + (long long)r * ldS;
    int rok = rowmask[r] > 0.f;
    float mx = -3.4e38f;
    for (int c = tid; c < cols; c += 256) {
        float v = (rok && colmask[c] > 0.f) ? Sr[c] : NEGV;
        mx = fmaxf(mx, v);
    }
    mx = block_reduce_max(mx, red);
    float sum = 0;
    for (int c = tid; c < cols; c += 256) {
        float v = (rok && colmask[c] > 0.f) ? Sr[c] : NEGV;
        sum += __expf(v - mx);
    }
    sum = block_reduce_sum(sum, red);
    float inv = 1.f / sum;
    for (int c = tid; c < cols; c += 256) {
        float v = (rok && colmask[c] > 0.f) ? Sr[c] : NEGV;
        float pv = __expf(v - mx) * inv;
        if (Pb) Pb[(long long)h * sPb + (long long)r * ldS + c] = (bf16_t)pv;
        if (Pf) Pf[(long long)h * sPf + (long long)r * ldS + c] = pv;
    }
}

// global max of col sums (gmax[0]) and row sums (gmax[1]) of a2 (entries >= 0)
__global__ void a2_norm_max_kernel(const float* a2, float* gmax) {
    int t = blockIdx.x * 256 + threadIdx.x;
    if (t >= 2 * H * MM) return;
    int half = t / (H * MM), r = t % (H * MM);
    int h = r / MM, x = r % MM;
    const float* A = a2 + (long long)h * MM * MM;
    float s = 0;
    if (half == 0) { for (int i = 0; i < MM; ++i) s += A[(long long)i * MM + x];
                     atomicMax((int*)&gmax[0], __float_as_int(s)); }
    else           { for (int j = 0; j < MM; ++j) s += A[(long long)x * MM + j];
                     atomicMax((int*)&gmax[1], __float_as_int(s)); }
}

__global__ void pinv_init_kernel(const float* a2, float* z, const float* gmax) {
    long long n = (long long)H * MM * MM;
    float inv = 1.f / (gmax[0] * gmax[1]);
    for (long long t = blockIdx.x * 256LL + threadIdx.x; t < n;
         t += (long long)gridDim.x * 256) {
        int j = (int)(t % MM);
        long long r = t / MM;
        int i = (int)(r % MM);
        int h = (int)(r / MM);
        z[((long long)h * MM + i) * MM + j] = a2[((long long)h * MM + j) * MM + i] * inv;
    }
}

__global__ void diag_minus_kernel(const float* in, float* out, float coef) {
    long long n = (long long)H * MM * MM;
    for (long long t = blockIdx.x * 256LL + threadIdx.x; t < n;
         t += (long long)gridDim.x * 256) {
        int j = (int)(t % MM);
        int i = (int)((t / MM) % MM);
        out[t] = coef * (i == j ? 1.f : 0.f) - in[t];
    }
}

// zavT[h][d][m] = (bf16) zav[h][m][d]
__global__ void trans_zav_kernel(const float* zav, bf16_t* zavT) {
    long long n = (long long)H * MM * DH;
    for (long long t = blockIdx.x * 256LL + threadIdx.x; t < n;
         t += (long long)gridDim.x * 256) {
        int d = (int)(t % DH);
        long long r = t / DH;
        int m = (int)(r % MM);
        int h = (int)(r / MM);
        zavT[((long long)h * DH + d) * MM + m] = (bf16_t)zav[t];
    }
}

// depthwise conv (kernel along sequence) accumulated into interleaved attn out
__global__ void conv_add_kernel(const float* vF, const float* w, float* attnF) {
    long long n = (long long)H * NP * DH;
    for (long long t = blockIdx.x * 256LL + threadIdx.x; t < n;
         t += (long long)gridDim.x * 256) {
        int d = (int)(t % DH);
        long long r = t / DH;
        int i = (int)(r % NP);
        int h = (int)(r / NP);
        float acc = 0;
        for (int k = 0; k < KWC; ++k) {
            int s = i + k - KWC / 2;
            if (s >= 0 && s < NP)
                acc += w[h * KWC + k] * vF[((long long)h * NP + s) * DH + d];
        }
        attnF[(long long)i * E + h * DH + d] += acc;
    }
}

__global__ void assign_idx_kernel(const float* x1, int* idx) {
    __shared__ float red[256];
    __shared__ float dots[CC];
    int t = blockIdx.x, tid = threadIdx.x;
    const float* tok = x1 + (long long)(PAD1 + 9 + t) * E;
    float part[CC] = {0, 0, 0, 0, 0, 0, 0, 0};
    for (int e = tid; e < E; e += 256) {
        float tv = tok[e];
#pragma unroll
        for (int c = 0; c < CC; ++c) part[c] += tv * x1[(long long)(PAD1 + 1 + c) * E + e];
    }
    for (int c = 0; c < CC; ++c) {
        float s = block_reduce_sum(part[c], red);
        if (tid == 0) dots[c] = s;
    }
    __syncthreads();
    if (tid == 0) {
        int best = 0; float bv = dots[0];
        for (int c = 1; c < CC; ++c) if (dots[c] > bv) { bv = dots[c]; best = c; }
        idx[t] = best;
    }
}

__global__ void hist_kernel(const int* idx, int* counts) {
    int t = blockIdx.x * 256 + threadIdx.x;
    if (t < NTOK) atomicAdd(&counts[idx[t]], 1);
}

__global__ void build_seq0_kernel(const float* x1, float* seq0, int c) {
    int i = blockIdx.x, tid = threadIdx.x;
    float* dst = seq0 + (long long)i * E;
    if (i < PAD0) { for (int e = tid; e < E; e += 256) dst[e] = 0.f; return; }
    const float* src = (i == PAD0) ? (x1 + (long long)(PAD1 + 1 + c) * E)
                                   : (x1 + (long long)(PAD1 + 9 + (i - PAD0 - 1)) * E);
    for (int e = tid; e < E; e += 256) dst[e] = src[e];
}

__global__ void build_mask0_kernel(const int* idx, float* mask0, int c) {
    for (int i = blockIdx.x * 256 + threadIdx.x; i < NP; i += gridDim.x * 256) {
        float m;
        if (i < PAD0) m = 0.f;
        else if (i == PAD0) m = 1.f;
        else m = (idx[i - PAD0 - 1] == c) ? 1.f : 0.f;
        mask0[i] = m;
    }
}

__global__ void merge_acc_kernel(const float* xout0, const int* idx, float* mergeS, int c) {
    int e = blockIdx.x * 256 + threadIdx.x;
    if (e >= E) return;
    float s = 0;
    for (int t = 0; t < NTOK; ++t)
        if (idx[t] == c) s += xout0[(long long)(PAD0 + 1 + t) * E + e];
    mergeS[c * E + e] = s;
}

__global__ void finalize_kernel(const float* x1, const float* mergeS, const int* counts,
                                const float* g, const float* b, const float* W3,
                                const float* b3, float* out) {
    __shared__ float red[256];
    __shared__ float tok[E];
    __shared__ float sc[4];
    int tid = threadIdx.x;
    const float* cls = x1 + (long long)PAD1 * E;
    float s = 0;
    for (int e = tid; e < E; e += 256) s += cls[e];
    float mu = block_reduce_sum(s, red) * (1.f / E);
    float v = 0;
    for (int e = tid; e < E; e += 256) { float d = cls[e] - mu; v += d * d; }
    float var = block_reduce_sum(v, red) * (1.f / E);
    float rstd = rsqrtf(var + 1e-5f);
    for (int e = tid; e < E; e += 256) {
        float hn = (cls[e] - mu) * rstd * g[e] + b[e];
        float acc = hn;
        for (int c = 0; c < CC; ++c)
            acc += mergeS[c * E + e] / fmaxf((float)counts[c], 1.f);
        tok[e] = acc * (1.f / (CC + 1));
    }
    __syncthreads();
    float l0 = 0, l1 = 0;
    for (int e = tid; e < E; e += 256) { l0 += tok[e] * W3[e * 2]; l1 += tok[e] * W3[e * 2 + 1]; }
    l0 = block_reduce_sum(l0, red);
    l1 = block_reduce_sum(l1, red);
    if (tid == 0) { sc[0] = l0 + b3[0]; sc[1] = l1 + b3[1]; sc[2] = 0.f; }
    __syncthreads();
    for (int p = 0; p < CC * CC; ++p) {
        int i = p >> 3, j = p & 7;
        float d = 0;
        for (int e = tid; e < E; e += 256)
            d += x1[(long long)(PAD1 + 1 + i) * E + e] * x1[(long long)(PAD1 + 1 + j) * E + e];
        float dot = block_reduce_sum(d, red);
        if (tid == 0) { float df = dot - (i == j ? 1.f : 0.f); sc[2] += df * df; }
        __syncthreads();
    }
    if (tid == 0) {
        float L0 = sc[0], L1 = sc[1];
        float m = fmaxf(L0, L1);
        float e0 = __expf(L0 - m), e1 = __expf(L1 - m);
        out[0] = L0; out[1] = L1;
        out[2] = e0 / (e0 + e1); out[3] = e1 / (e0 + e1);
        out[4] = sqrtf(sc[2]);
    }
}

// ---------------------------------------------------------------------------
// host-side launch helpers
// ---------------------------------------------------------------------------
static void launch_gemm_bf16(hipStream_t st, const bf16_t* A, const bf16_t* Bt,
                             int M, int N, int K, int lda, int ldb,
                             float* Cf, bf16_t* Cb, int ldc,
                             const float* bias, const float* resid, int ldres, int relu,
                             long long sA, long long sB, long long sC, long long sRes,
                             int batch) {
    dim3 g((N + 127) / 128, M / 16, batch), blk(128);
    hipLaunchKernelGGL(gemm_bf16_kernel, g, blk, 0, st, A, Bt, M, N, K, lda, ldb,
                       Cf, Cb, ldc, bias, resid, ldres, relu, sA, sB, sC, sRes);
}
static void launch_gemm_f32(hipStream_t st, const float* A, const float* B,
                            int M, int N, int K, int lda, int ldb,
                            float* Cf, int ldc, float dA, float s,
                            long long sA, long long sB, long long sC, int batch) {
    dim3 g((N + 127) / 128, M / 16, batch), blk(128);
    hipLaunchKernelGGL(gemm_f32_kernel, g, blk, 0, st, A, B, M, N, K, lda, ldb,
                       Cf, ldc, dA, s, sA, sB, sC);
}

struct AttnBufs {
    bf16_t* lnB; float* qkvF;
    float *qF, *kF, *vF; bf16_t *qB, *kB, *vTB;
    float* mls; bf16_t *qlB, *klB;
    float* simF; bf16_t* a1B; bf16_t* a3B;
    float* sim2F; float* a2F;
    float *zF, *t0, *t1, *t2; float* gmax;
    float* a3vF; float* zavF; bf16_t* zavTB;
    float* attnF; bf16_t* attnB;
};

static void run_attn(hipStream_t st, const float* seq, const float* mask, int padRows,
                     const bf16_t* WqkvTB, const bf16_t* WoutTB, const float* bout,
                     const float* convw, const float* lng, const float* lnb,
                     float* xout, AttnBufs& B) {
    const long long sHNP = (long long)NP * DH;   // per-head q/k/v stride
    const long long sHM  = (long long)MM * DH;   // per-head landmark stride
    const long long sS1  = (long long)NP * MM;   // per-head sim1/a1 stride
    const long long sS3  = (long long)MM * NP;   // per-head sim3/a3 stride
    const long long sMM  = (long long)MM * MM;

    // 1. LayerNorm of real rows (padded rows -> 0)
    hipLaunchKernelGGL(layernorm_kernel, dim3(NP), dim3(256), 0, st, seq, B.lnB, lng, lnb, padRows);
    // 2. qkv = ln @ Wqkv      (B^T = WqkvT, (3E x E))
    launch_gemm_bf16(st, B.lnB, WqkvTB, NP, 3 * E, E, E, E, B.qkvF, nullptr, 3 * E,
                     nullptr, nullptr, 0, 0, 0, 0, 0, 0, 1);
    // 3. split + mask + scale (also writes v transposed for the a3@v GEMM)
    hipLaunchKernelGGL(qkv_split_kernel, dim3(2048), dim3(256), 0, st, B.qkvF, mask,
                       B.qF, B.kF, B.vF, B.qB, B.kB, B.vTB);
    // 4. landmarks
    hipLaunchKernelGGL(mls_kernel, dim3(1), dim3(256), 0, st, mask, B.mls);
    hipLaunchKernelGGL(landmark_kernel, dim3(MM, H), dim3(DH), 0, st, B.qF, B.kF, B.mls,
                       B.qlB, B.klB);
    // 5. sim1 = q @ kl^T (kl already N x K), softmax -> a1 (bf16)
    launch_gemm_bf16(st, B.qB, B.klB, NP, MM, DH, DH, DH, B.simF, nullptr, MM,
                     nullptr, nullptr, 0, 0, sHNP, sHM, sS1, 0, H);
    hipLaunchKernelGGL(softmax_mask_kernel, dim3(NP, H), dim3(256), 0, st, B.simF, B.a1B,
                       (float*)nullptr, mask, B.mls, MM, MM, sS1, sS1, 0LL);
    // 6. sim2 = ql @ kl^T, softmax -> a2 (f32)
    launch_gemm_bf16(st, B.qlB, B.klB, MM, MM, DH, DH, DH, B.sim2F, nullptr, MM,
                     nullptr, nullptr, 0, 0, sHM, sHM, sMM, 0, H);
    hipLaunchKernelGGL(softmax_mask_kernel, dim3(MM, H), dim3(256), 0, st, B.sim2F,
                       (bf16_t*)nullptr, B.a2F, B.mls, B.mls, MM, MM, sMM, 0LL, sMM);
    // 7. sim3 = ql @ k^T (k already N x K), softmax -> a3 (bf16)
    launch_gemm_bf16(st, B.qlB, B.kB, MM, NP, DH, DH, DH, B.simF, nullptr, NP,
                     nullptr, nullptr, 0, 0, sHM, sHNP, sS3, 0, H);
    hipLaunchKernelGGL(softmax_mask_kernel, dim3(MM, H), dim3(256), 0, st, B.simF, B.a3B,
                       (float*)nullptr, B.mls, mask, NP, NP, sS3, sS3, 0LL);
    // 8. a3v = a3 @ v   (B^T = vT, (DH x NP)); f32 out feeds f32 WMMA
    launch_gemm_bf16(st, B.a3B, B.vTB, MM, DH, NP, NP, NP, B.a3vF, nullptr, DH,
                     nullptr, nullptr, 0, 0, sS3, sHNP, sHM, 0, H);
    // 9. pinv(a2): 6 Newton-Schulz iterations in fp32 WMMA
    hipMemsetAsync(B.gmax, 0, 2 * sizeof(float), st);
    hipLaunchKernelGGL(a2_norm_max_kernel, dim3((2 * H * MM + 255) / 256), dim3(256), 0, st,
                       B.a2F, B.gmax);
    hipLaunchKernelGGL(pinv_init_kernel, dim3(2048), dim3(256), 0, st, B.a2F, B.zF, B.gmax);
    float* zcur = B.zF;
    float* tswap = B.t2;
    for (int it = 0; it < 6; ++it) {
        launch_gemm_f32(st, B.a2F, zcur, MM, MM, MM, MM, MM, B.t0, MM, 0.f, 1.f,
                        sMM, sMM, sMM, H);                                       // xz
        hipLaunchKernelGGL(diag_minus_kernel, dim3(2048), dim3(256), 0, st, B.t0, B.t1, 7.f);
        launch_gemm_f32(st, B.t0, B.t1, MM, MM, MM, MM, MM, tswap, MM, 15.f, -1.f,
                        sMM, sMM, sMM, H);                                       // 15I - xz@w1
        launch_gemm_f32(st, B.t0, tswap, MM, MM, MM, MM, MM, B.t1, MM, 13.f, -1.f,
                        sMM, sMM, sMM, H);                                       // 13I - xz@w2
        launch_gemm_f32(st, zcur, B.t1, MM, MM, MM, MM, MM, tswap, MM, 0.f, 0.25f,
                        sMM, sMM, sMM, H);                                       // z_new
        float* tmp = zcur; zcur = tswap; tswap = tmp;
    }
    // 10. zav = z @ a3v  (f32 WMMA), then tiny transpose to (DH x MM) bf16
    launch_gemm_f32(st, zcur, B.a3vF, MM, DH, MM, MM, DH, B.zavF, DH, 0.f, 1.f,
                    sMM, sHM, sHM, H);
    hipLaunchKernelGGL(trans_zav_kernel, dim3(128), dim3(256), 0, st, B.zavF, B.zavTB);
    // 11. out = a1 @ zav  -> interleaved (NP x E) with per-head column offset
    launch_gemm_bf16(st, B.a1B, B.zavTB, NP, DH, MM, MM, MM, B.attnF, nullptr, E,
                     nullptr, nullptr, 0, 0, sS1, sHM, (long long)DH, 0, H);
    // 12. += depthwise conv(v)
    hipLaunchKernelGGL(conv_add_kernel, dim3(2048), dim3(256), 0, st, B.vF, convw, B.attnF);
    hipLaunchKernelGGL(cvt_kernel, dim3(2048), dim3(256), 0, st, B.attnF, B.attnB,
                       (long long)NP * E);
    // 13. xout = attn @ Wout + bout + seq (residual)   (B^T = WoutT)
    launch_gemm_bf16(st, B.attnB, WoutTB, NP, E, E, E, E, xout, nullptr, E,
                     bout, seq, E, 0, 0, 0, 0, 0, 1);
}

// ---------------------------------------------------------------------------
extern "C" void kernel_launch(void* const* d_in, const int* in_sizes, int n_in,
                              void* d_out, int out_size, void* d_ws, size_t ws_size,
                              hipStream_t stream) {
    (void)in_sizes; (void)n_in; (void)out_size; (void)ws_size;
    const float* h     = (const float*)d_in[0];
    const float* W1    = (const float*)d_in[1];
    const float* b1    = (const float*)d_in[2];
    const float* cls   = (const float*)d_in[3];
    const float* pmt   = (const float*)d_in[4];
    const float* ln1g  = (const float*)d_in[5];
    const float* ln1b  = (const float*)d_in[6];
    const float* Wqkv1 = (const float*)d_in[7];
    const float* Wout1 = (const float*)d_in[8];
    const float* bout1 = (const float*)d_in[9];
    const float* conv1 = (const float*)d_in[10];
    const float* ln0g  = (const float*)d_in[11];
    const float* ln0b  = (const float*)d_in[12];
    const float* Wqkv0 = (const float*)d_in[13];
    const float* Wout0 = (const float*)d_in[14];
    const float* bout0 = (const float*)d_in[15];
    const float* conv0 = (const float*)d_in[16];
    const float* lnfg  = (const float*)d_in[17];
    const float* lnfb  = (const float*)d_in[18];
    const float* W3    = (const float*)d_in[19];
    const float* b3    = (const float*)d_in[20];
    float* outp = (float*)d_out;

    char* p = (char*)d_ws;
    auto alloc = [&](size_t bytes) -> char* {
        char* r = p; p += (bytes + 255) & ~(size_t)255; return r;
    };
    // bf16 copies: activations row-major; weights transposed (N x K)
    bf16_t* hB      = (bf16_t*)alloc((size_t)NTOK * IND * 2);
    bf16_t* W1TB    = (bf16_t*)alloc((size_t)IND * E * 2);
    bf16_t* Wqkv1TB = (bf16_t*)alloc((size_t)E * 3 * E * 2);
    bf16_t* Wout1TB = (bf16_t*)alloc((size_t)E * E * 2);
    bf16_t* Wqkv0TB = (bf16_t*)alloc((size_t)E * 3 * E * 2);
    bf16_t* Wout0TB = (bf16_t*)alloc((size_t)E * E * 2);
    // sequences / masks
    float* seqpad = (float*)alloc((size_t)NP * E * 4);
    float* x1     = (float*)alloc((size_t)NP * E * 4);
    float* seq0   = (float*)alloc((size_t)NP * E * 4);
    float* xout0  = (float*)alloc((size_t)NP * E * 4);
    float* mask1  = (float*)alloc((size_t)NP * 4);
    float* mask0  = (float*)alloc((size_t)NP * 4);
    // attention scratch
    AttnBufs B;
    B.lnB   = (bf16_t*)alloc((size_t)NP * E * 2);
    B.qkvF  = (float*)alloc((size_t)NP * 3 * E * 4);
    B.qF    = (float*)alloc((size_t)H * NP * DH * 4);
    B.kF    = (float*)alloc((size_t)H * NP * DH * 4);
    B.vF    = (float*)alloc((size_t)H * NP * DH * 4);
    B.qB    = (bf16_t*)alloc((size_t)H * NP * DH * 2);
    B.kB    = (bf16_t*)alloc((size_t)H * NP * DH * 2);
    B.vTB   = (bf16_t*)alloc((size_t)H * NP * DH * 2);
    B.mls   = (float*)alloc((size_t)MM * 4);
    B.qlB   = (bf16_t*)alloc((size_t)H * MM * DH * 2);
    B.klB   = (bf16_t*)alloc((size_t)H * MM * DH * 2);
    B.simF  = (float*)alloc((size_t)H * NP * MM * 4);
    B.a1B   = (bf16_t*)alloc((size_t)H * NP * MM * 2);
    B.a3B   = (bf16_t*)alloc((size_t)H * MM * NP * 2);
    B.sim2F = (float*)alloc((size_t)H * MM * MM * 4);
    B.a2F   = (float*)alloc((size_t)H * MM * MM * 4);
    B.zF    = (float*)alloc((size_t)H * MM * MM * 4);
    B.t0    = (float*)alloc((size_t)H * MM * MM * 4);
    B.t1    = (float*)alloc((size_t)H * MM * MM * 4);
    B.t2    = (float*)alloc((size_t)H * MM * MM * 4);
    B.gmax  = (float*)alloc(16);
    B.a3vF  = (float*)alloc((size_t)H * MM * DH * 4);
    B.zavF  = (float*)alloc((size_t)H * MM * DH * 4);
    B.zavTB = (bf16_t*)alloc((size_t)H * MM * DH * 2);
    B.attnF = (float*)alloc((size_t)NP * E * 4);
    B.attnB = (bf16_t*)alloc((size_t)NP * E * 2);
    // clustering state
    int*   idx    = (int*)alloc((size_t)NTOK * 4);
    int*   counts = (int*)alloc((size_t)CC * 4);
    float* mergeS = (float*)alloc((size_t)CC * E * 4);

    // ---- weight conversions (weights transposed into B^T layout) ----
    hipLaunchKernelGGL(cvt_kernel, dim3(2048), dim3(256), 0, stream, h, hB, (long long)NTOK * IND);
    hipLaunchKernelGGL(cvt_trans_kernel, dim3(1024), dim3(256), 0, stream, W1, W1TB, IND, E);
    hipLaunchKernelGGL(cvt_trans_kernel, dim3(1024), dim3(256), 0, stream, Wqkv1, Wqkv1TB, E, 3 * E);
    hipLaunchKernelGGL(cvt_trans_kernel, dim3(512),  dim3(256), 0, stream, Wout1, Wout1TB, E, E);
    hipLaunchKernelGGL(cvt_trans_kernel, dim3(1024), dim3(256), 0, stream, Wqkv0, Wqkv0TB, E, 3 * E);
    hipLaunchKernelGGL(cvt_trans_kernel, dim3(512),  dim3(256), 0, stream, Wout0, Wout0TB, E, E);

    // ---- block 1: build padded seq = [pad | cls | pmt | relu(h@W1+b1)] ----
    hipMemsetAsync(seqpad, 0, (size_t)NP * E * 4, stream);
    hipLaunchKernelGGL(build_seq1_kernel, dim3(9), dim3(256), 0, stream, cls, pmt, seqpad);
    launch_gemm_bf16(stream, hB, W1TB, NTOK, E, IND, IND, IND,
                     seqpad + (long long)(PAD1 + 9) * E, nullptr, E,
                     b1, nullptr, 0, 1, 0, 0, 0, 0, 1);
    hipLaunchKernelGGL(set_mask1_kernel, dim3(32), dim3(256), 0, stream, mask1);
    run_attn(stream, seqpad, mask1, PAD1, Wqkv1TB, Wout1TB, bout1, conv1, ln1g, ln1b, x1, B);

    // ---- cluster assignment ----
    hipLaunchKernelGGL(assign_idx_kernel, dim3(NTOK), dim3(256), 0, stream, x1, idx);
    hipMemsetAsync(counts, 0, CC * 4, stream);
    hipLaunchKernelGGL(hist_kernel, dim3(NTOK / 256), dim3(256), 0, stream, idx, counts);

    // ---- per-cluster masked attention, accumulate merge sums ----
    for (int c = 0; c < CC; ++c) {
        hipLaunchKernelGGL(build_seq0_kernel, dim3(NP), dim3(256), 0, stream, x1, seq0, c);
        hipLaunchKernelGGL(build_mask0_kernel, dim3(32), dim3(256), 0, stream, idx, mask0, c);
        run_attn(stream, seq0, mask0, PAD0, Wqkv0TB, Wout0TB, bout0, conv0, ln0g, ln0b, xout0, B);
        hipLaunchKernelGGL(merge_acc_kernel, dim3(2), dim3(256), 0, stream, xout0, idx, mergeS, c);
    }

    // ---- head: LN(cls) + cluster means -> logits, softmax, ptc ----
    hipLaunchKernelGGL(finalize_kernel, dim3(1), dim3(256), 0, stream,
                       x1, mergeS, counts, lnfg, lnfb, W3, b3, outp);
}
